// LIFT_18451179503779
// MI455X (gfx1250) — matrix operand order repaired and checked by reading, not compiled, not run on hardware
//
#include <hip/hip_runtime.h>
#include <math.h>

// ---------------------------------------------------------------------------
// Problem constants (from reference): B=16, L=720, C=128, H=96, K=8, S=8
// F2 = 49, OUT = F2*(2K+1) = 833, NB = B*C = 2048 rows.
// Spectral dim for rfft(720): 361 complex -> 722 reals, padded to 736 (23*32).
// ---------------------------------------------------------------------------
#define NB    2048
#define LDIM  720
#define KPAD  736
#define ALD   744     // LDS leading dim for A tiles (bank-conflict-free)
#define HDIM  96
#define F2    49
#define OUTF  833

typedef __attribute__((ext_vector_type(16))) __bf16 v16bf;
typedef __attribute__((ext_vector_type(8)))  float  v8f;
typedef __attribute__((ext_vector_type(4)))  int    v4i;

// ---------------------------------------------------------------------------
// Async global->LDS copy (CDNA5 ASYNCcnt path), guarded so compile never fails.
// Builtin signature (from hipcc diagnostic): (v4i* src, v4i* dst, imm, imm),
// generic pointers — backend infers AS(1)/AS(3) from the pointer origins.
// ---------------------------------------------------------------------------
#if defined(__HIP_DEVICE_COMPILE__) && \
    __has_builtin(__builtin_amdgcn_global_load_async_to_lds_b128) && \
    __has_builtin(__builtin_amdgcn_s_wait_asynccnt)
#define ASYNC_LDS 1
#endif

__device__ __forceinline__ void async_copy16(const void* g, void* l) {
#ifdef ASYNC_LDS
    __builtin_amdgcn_global_load_async_to_lds_b128((v4i*)g, (v4i*)l, 0, 0);
#else
    *(float4*)l = *(const float4*)g;
#endif
}

__device__ __forceinline__ void async_wait0() {
#ifdef ASYNC_LDS
    __builtin_amdgcn_s_wait_asynccnt(0);
#endif
}

__device__ __forceinline__ v8f zero_v8f() {
    v8f z;
#pragma unroll
    for (int i = 0; i < 8; ++i) z[i] = 0.0f;
    return z;
}

// A fragment: 16x32 bf16, element (m,k) at base[(m0+m)*ld + k0+k]  (row-major).
// Per ISA 7.12.2: lane<16 -> M=lane, K in {0..7}U{16..23};
//                 lane>=16 -> M=lane-16, K in {8..15}U{24..31}.
__device__ __forceinline__ v16bf load_a_frag(const __bf16* base, int ld,
                                             int m0, int k0, int lane) {
    int m     = lane & 15;
    int kbase = (lane < 16) ? 0 : 8;
    union { v16bf v; __bf16 e[16]; } u;
#pragma unroll
    for (int v = 0; v < 8; ++v) {
        int k = ((v >> 2) << 4) + kbase + ((v & 3) << 1);
        const __bf16* p = base + (size_t)(m0 + m) * ld + (k0 + k);
        u.e[2 * v]     = p[0];
        u.e[2 * v + 1] = p[1];
    }
    return u.v;
}

// B fragment: 32x16 bf16, element (k,n) stored K-contiguous (transposed):
// baseT[(n0+n)*ld + k0+k].  lanes 0-15: N=lane, K in {0..7}U{16..23}; lanes 16-31: K in {8..15}U{24..31}.
__device__ __forceinline__ v16bf load_b_frag(const __bf16* baseT, int ld,
                                             int n0, int k0, int lane) {
    int n  = lane & 15;
    int kb = (lane < 16) ? 0 : 8;
    union { v16bf v; __bf16 e[16]; } u;
#pragma unroll
    for (int v = 0; v < 8; ++v) {
        int k = ((v >> 2) << 4) + kb + ((v & 3) << 1);
        const __bf16* p = baseT + (size_t)(n0 + n) * ld + (k0 + k);
        u.e[2 * v]     = p[0];
        u.e[2 * v + 1] = p[1];
    }
    return u.v;
}

// ---------------------------------------------------------------------------
// K0: build DFT tables.  Integer-mod angle reduction keeps twiddles exact.
//   BfT[f][l] : forward rfft(720) coeff (transposed, K=l contiguous)
//   BiT[lag][f]: inverse coeff incl. conj-sym doubling and 1/L^2 scaling
//   c96/s96   : rfft(96)/irfft(96) twiddles
// ---------------------------------------------------------------------------
__global__ void k_tables(__bf16* BfT, __bf16* BiT, float* c96, float* s96) {
    const float PI2 = 6.28318530717958647692f;
    const int t1 = KPAD * KPAD;      // BfT: 736 x 736
    const int t2 = LDIM * KPAD;      // BiT: 720 x 736
    const int t3 = F2 * HDIM;        // twiddles
    for (int t = blockIdx.x * blockDim.x + threadIdx.x; t < t1 + t2 + t3;
         t += gridDim.x * blockDim.x) {
        if (t < t1) {
            int f = t / KPAD, l = t - f * KPAD;
            float v = 0.0f;
            if (l < LDIM && f < 722) {
                if (f < 361) { int m = (f * l) % LDIM; v = cosf(PI2 * m * (1.0f / LDIM)); }
                else { int fr = f - 361; int m = (fr * l) % LDIM; v = -sinf(PI2 * m * (1.0f / LDIM)); }
            }
            BfT[t] = (__bf16)v;
        } else if (t < t1 + t2) {
            int u = t - t1;
            int lag = u / KPAD, f = u - lag * KPAD;
            float v = 0.0f;
            const float sc = 1.0f / (720.0f * 720.0f);   // irfft 1/L and reference /L
            if (f < 361) {
                int m = (f * lag) % LDIM;
                float c = (f == 0 || f == 360) ? 1.0f : 2.0f;
                v = c * sc * cosf(PI2 * m * (1.0f / LDIM));
            } else if (f < 722) {
                int fr = f - 361;
                int m = (fr * lag) % LDIM;
                float c = (fr == 0 || fr == 360) ? 1.0f : 2.0f;
                v = -c * sc * sinf(PI2 * m * (1.0f / LDIM));
            }
            BiT[u] = (__bf16)v;
        } else {
            int u = t - t1 - t2;
            int f = u / HDIM, h = u - f * HDIM;
            int m = (f * h) % HDIM;
            c96[u] = cosf(PI2 * m * (1.0f / HDIM));
            s96[u] = sinf(PI2 * m * (1.0f / HDIM));
        }
    }
}

// ---------------------------------------------------------------------------
// K1: transpose + normalize.  One block per (b,c) row.
// ---------------------------------------------------------------------------
__global__ __launch_bounds__(256) void k_norm(const float* __restrict__ x,
                                              const float* __restrict__ yh,
                                              float* xn, float* yn, __bf16* xnb,
                                              float* muA, float* stdA) {
    int bc = blockIdx.x;
    int b = bc >> 7, c = bc & 127;
    int tid = threadIdx.x;
    __shared__ float rs[256], rs2[256];
    __shared__ float smu, sisd;
    float s = 0.0f, s2 = 0.0f;
    for (int l = tid; l < LDIM; l += 256) {
        float v = x[((size_t)b * LDIM + l) * 128 + c];
        s += v; s2 += v * v;
    }
    rs[tid] = s; rs2[tid] = s2;
    __syncthreads();
    for (int off = 128; off > 0; off >>= 1) {
        if (tid < off) { rs[tid] += rs[tid + off]; rs2[tid] += rs2[tid + off]; }
        __syncthreads();
    }
    if (tid == 0) {
        float mu  = rs[0] * (1.0f / LDIM);
        float var = rs2[0] * (1.0f / LDIM) - mu * mu;
        float sd  = sqrtf(var + 1e-8f);
        smu = mu; sisd = 1.0f / sd;
        muA[bc] = mu; stdA[bc] = sd;
    }
    __syncthreads();
    float mu = smu, isd = sisd;
    for (int l = tid; l < LDIM; l += 256) {
        float v = (x[((size_t)b * LDIM + l) * 128 + c] - mu) * isd;
        xn[(size_t)bc * LDIM + l]  = v;
        xnb[(size_t)bc * KPAD + l] = (__bf16)v;
    }
    for (int l = LDIM + tid; l < KPAD; l += 256) xnb[(size_t)bc * KPAD + l] = (__bf16)0.0f;
    for (int h = tid; h < HDIM; h += 256) {
        float v = (yh[((size_t)b * HDIM + h) * 128 + c] - mu) * isd;
        yn[(size_t)bc * HDIM + h] = v;
    }
}

// ---------------------------------------------------------------------------
// K2: rfft(720) as WMMA GEMM:  RF[2048 x 736] = xnb[2048 x 736] * BfT^T
// The 16x736 bf16 A-tile is async-staged into LDS ONCE per block and shared
// by all 8 waves (removes 7/8 of the A global traffic).  B streams from L2.
// ---------------------------------------------------------------------------
__global__ __launch_bounds__(256) void k_rfft(const __bf16* __restrict__ xnb,
                                              const __bf16* __restrict__ BfT,
                                              float* __restrict__ RF) {
    __shared__ __align__(16) __bf16 Ash[16 * ALD];
    int m0  = blockIdx.x * 16;
    int tid = threadIdx.x;

    const __bf16* src = xnb + (size_t)m0 * KPAD;
    for (int t = tid; t < 16 * (KPAD / 8); t += 256) {      // 16B chunks
        int r = t / (KPAD / 8), ch = t - r * (KPAD / 8);
        async_copy16(src + (size_t)r * KPAD + ch * 8, Ash + r * ALD + ch * 8);
    }
    async_wait0();
    __syncthreads();

    int wave = tid >> 5;
    int lane = tid & 31;
    for (int nt = wave; nt < KPAD / 16; nt += 8) {
        int n0 = nt * 16;
        v8f acc = zero_v8f();
        for (int kt = 0; kt < KPAD / 32; ++kt) {
            v16bf a  = load_a_frag(Ash, ALD, 0, kt * 32, lane);
            v16bf bm = load_b_frag(BfT, KPAD, n0, kt * 32, lane);
            acc = __builtin_amdgcn_wmma_f32_16x16x32_bf16(false, a, false, bm,
                                                          (short)0, acc, false, false);
        }
        int n     = n0 + (lane & 15);
        int mbase = m0 + ((lane < 16) ? 0 : 8);
#pragma unroll
        for (int r = 0; r < 8; ++r)
            RF[(size_t)(mbase + r) * KPAD + n] = acc[r];
    }
}

// ---------------------------------------------------------------------------
// K3: pair cross-spectrum -> inverse-DFT GEMM (WMMA) -> fused peak/argmax.
// One block = (b, i, 16 consecutive j).  RF row of channel i is async-staged
// into LDS (reused 16x); cross-spectrum tile built in LDS (bf16); correlation
// rows produced in two 368-lag passes so CC + A fit under 64 KB static LDS.
// ---------------------------------------------------------------------------
__global__ __launch_bounds__(128) void k_corr(const float* __restrict__ RF,
                                              const __bf16* __restrict__ BiT,
                                              float* cmaxA, float* rvalA, int* shiftA) {
    int j0 = blockIdx.x * 16;
    int i  = blockIdx.y;
    int b  = blockIdx.z;
    int tid  = threadIdx.x;
    int wave = tid >> 5, lane = tid & 31;

    __shared__ __align__(16) __bf16 Ash[16][ALD];   // cross-spectrum tile (bf16)
    __shared__ __align__(16) float  RFis[KPAD];     // RF row of channel i
    __shared__ float  CC[16][372];                  // correlation rows, one pass
    __shared__ float  s_ba[128], s_bv[128];
    __shared__ int    s_bl[128];

    const float* RFi = RF + (size_t)(b * 128 + i) * KPAD;
    for (int t = tid; t < KPAD / 4; t += 128)       // 736 floats, 16B chunks
        async_copy16(RFi + t * 4, RFis + t * 4);
    async_wait0();
    __syncthreads();

    // Build A tile: A[j][f] = rf_i * conj(rf_j), layout [Re0..Re360, Im0..Im360].
    for (int t = tid; t < 16 * KPAD; t += 128) {
        int j = t / KPAD, f = t - j * KPAD;
        const float* RFj = RF + (size_t)(b * 128 + j0 + j) * KPAD;
        float v;
        if (f < 361)      v = RFis[f] * RFj[f] + RFis[361 + f] * RFj[361 + f];
        else if (f < 722) { int fr = f - 361; v = RFis[361 + fr] * RFj[fr] - RFis[fr] * RFj[361 + fr]; }
        else              v = 0.0f;
        Ash[j][f] = (__bf16)v;
    }
    __syncthreads();

    int row = tid & 15, slot = tid >> 4;           // 8 reducer slots per row
    float bA = -1.0f, bV = 0.0f;
    int   bL = 0;

    for (int pass = 0; pass < 2; ++pass) {
        int base = pass * 352;                     // pass0: lags 0..367, pass1: 352..719
        for (int nt = wave; nt < 23; nt += 4) {
            int lag0 = base + nt * 16;
            __builtin_prefetch(BiT + (size_t)lag0 * KPAD, 0, 0);
            v8f acc = zero_v8f();
            for (int kt = 0; kt < KPAD / 32; ++kt) {
                v16bf a  = load_a_frag(&Ash[0][0], ALD, 0, kt * 32, lane);
                v16bf bm = load_b_frag(BiT, KPAD, lag0, kt * 32, lane);
                acc = __builtin_amdgcn_wmma_f32_16x16x32_bf16(false, a, false, bm,
                                                              (short)0, acc, false, false);
            }
            int n  = lane & 15;
            int jb = (lane < 16) ? 0 : 8;
#pragma unroll
            for (int r = 0; r < 8; ++r)
                CC[jb + r][(lag0 - base) + n] = acc[r];
        }
        __syncthreads();
        // candidates: trimmed lags; mask = local-max of |cc| (neighbors in-range)
        int lo = pass ? 367 : 1;
        int hi = pass ? 718 : 366;
        for (int l = lo + slot; l <= hi; l += 8) {
            int ci = l - base;
            float c0 = CC[row][ci];
            float a1 = fabsf(c0);
            bool pk = (a1 >= fabsf(CC[row][ci - 1])) && (a1 >= fabsf(CC[row][ci + 1]));
            float av = pk ? a1 : 0.0f;
            float vv = pk ? c0 : 0.0f;
            int lt = l - 1;                        // trimmed index
            if (av > bA || (av == bA && lt < bL)) { bA = av; bV = vv; bL = lt; }
        }
        __syncthreads();
    }
    s_ba[tid] = bA; s_bv[tid] = bV; s_bl[tid] = bL;
    __syncthreads();
    if (tid < 16) {
        float ba = s_ba[tid], bv = s_bv[tid];
        int   bl = s_bl[tid];
        for (int s2 = 1; s2 < 8; ++s2) {
            int t = tid + 16 * s2;
            float oa = s_ba[t];
            if (oa > ba || (oa == ba && s_bl[t] < bl)) { ba = oa; bv = s_bv[t]; bl = s_bl[t]; }
        }
        size_t o = ((size_t)b * 128 + i) * 128 + (j0 + tid);
        cmaxA[o]  = ba;
        rvalA[o]  = bv;
        shiftA[o] = bl + 1;                         // reference: shift + 1
    }
}

// ---------------------------------------------------------------------------
// K4: top-K leaders per (b,i) + softmax correlation weights.  One wave each.
// ---------------------------------------------------------------------------
__global__ __launch_bounds__(32) void k_topk(const float* __restrict__ cmaxA,
                                             const float* __restrict__ rvalA,
                                             const int* __restrict__ shiftA,
                                             const float* __restrict__ temperature,
                                             int* leadA, int* shSelA, float* sgnA, float* cfA) {
    int bi   = blockIdx.x;
    int lane = threadIdx.x;
    const float* cm = cmaxA + (size_t)bi * 128;
    float v[4]; int tk[4] = {0, 0, 0, 0};
#pragma unroll
    for (int q = 0; q < 4; ++q) v[q] = cm[lane * 4 + q];
    int lead_l[8];
    for (int k = 0; k < 8; ++k) {
        float mv = -1e30f; int mi = 1 << 30;
#pragma unroll
        for (int q = 0; q < 4; ++q) {
            if (!tk[q]) {
                int idx = lane * 4 + q;
                if (v[q] > mv || (v[q] == mv && idx < mi)) { mv = v[q]; mi = idx; }
            }
        }
        for (int off = 16; off > 0; off >>= 1) {
            float ov = __shfl_xor(mv, off, 32);
            int   oi = __shfl_xor(mi, off, 32);
            if (ov > mv || (ov == mv && oi < mi)) { mv = ov; mi = oi; }
        }
        if ((mi >> 2) == lane) tk[mi & 3] = 1;
        lead_l[k] = mi;
    }
    if (lane == 0) {
        float T = temperature[0];
        float lg[9];
        lg[0] = 1.0f / T;
        for (int k = 0; k < 8; ++k) {
            int j = lead_l[k];
            float r = rvalA[(size_t)bi * 128 + j];
            float ra = fabsf(r);
            lg[k + 1] = ra / T;
            leadA[bi * 8 + k]  = j;
            shSelA[bi * 8 + k] = shiftA[(size_t)bi * 128 + j];
            sgnA[bi * 8 + k]   = (r > 0.0f) ? 1.0f : ((r < 0.0f) ? -1.0f : 0.0f);
        }
        float mx = lg[0];
        for (int q = 1; q < 9; ++q) mx = fmaxf(mx, lg[q]);
        float e[9], den = 0.0f;
        for (int q = 0; q < 9; ++q) { e[q] = expf(lg[q] - mx); den += e[q]; }
        float iden = 1.0f / den;
        for (int k = 0; k < 8; ++k) cfA[bi * 8 + k] = e[k + 1] * iden;   // rw[1:]
    }
}

// ---------------------------------------------------------------------------
// K5: p = softmax(state_bias + basic_state + xt . cls_w^T)
// ---------------------------------------------------------------------------
__global__ __launch_bounds__(128) void k_p(const float* __restrict__ x,
                                           const float* __restrict__ cls_w,
                                           const float* __restrict__ basic_state,
                                           const float* __restrict__ state_bias,
                                           float* pA) {
    int bc = blockIdx.x;
    int b = bc >> 7, c = bc & 127;
    int tid = threadIdx.x;
    float acc[8];
#pragma unroll
    for (int s = 0; s < 8; ++s) acc[s] = 0.0f;
    for (int l = tid; l < LDIM; l += 128) {
        float xv = x[((size_t)b * LDIM + l) * 128 + c];
#pragma unroll
        for (int s = 0; s < 8; ++s) acc[s] += xv * cls_w[s * LDIM + l];
    }
    __shared__ float red[128];
    __shared__ float lg[8];
    for (int s = 0; s < 8; ++s) {
        red[tid] = acc[s];
        __syncthreads();
        for (int off = 64; off > 0; off >>= 1) {
            if (tid < off) red[tid] += red[tid + off];
            __syncthreads();
        }
        if (tid == 0) lg[s] = red[0] + state_bias[s] + basic_state[c * 8 + s];
        __syncthreads();
    }
    if (tid == 0) {
        float mx = lg[0];
        for (int s = 1; s < 8; ++s) mx = fmaxf(mx, lg[s]);
        float e[8], den = 0.0f;
        for (int s = 0; s < 8; ++s) { e[s] = expf(lg[s] - mx); den += e[s]; }
        float iden = 1.0f / den;
        for (int s = 0; s < 8; ++s) pA[bc * 8 + s] = e[s] * iden;
    }
}

// ---------------------------------------------------------------------------
// K6: filt[b,c,o] = sum_s p_s * ( mix_head_b[s,o] + sum_k cf_k * W[s,k*OUT+o] )
// ---------------------------------------------------------------------------
__global__ __launch_bounds__(256) void k_filt(const float* __restrict__ pA,
                                              const float* __restrict__ cfA,
                                              const float* __restrict__ mhw,
                                              const float* __restrict__ mhb,
                                              float* filt) {
    int bc = blockIdx.x;
    int tid = threadIdx.x;
    __shared__ float ps[8], cs[8];
    if (tid < 8)       ps[tid]     = pA[bc * 8 + tid];
    else if (tid < 16) cs[tid - 8] = cfA[bc * 8 + (tid - 8)];
    __syncthreads();
    for (int o = tid; o < OUTF; o += 256) {
        float acc = 0.0f;
#pragma unroll
        for (int s = 0; s < 8; ++s) {
            float t = mhb[s * OUTF + o];
#pragma unroll
            for (int k = 0; k < 8; ++k) t += cs[k] * mhw[(size_t)s * (8 * OUTF) + k * OUTF + o];
            acc += ps[s] * t;
        }
        filt[(size_t)bc * OUTF + o] = acc;
    }
}

// ---------------------------------------------------------------------------
// K7: per-(b,c) spectral filtering + 49x147 complex mix + irfft(96) + denorm.
// ---------------------------------------------------------------------------
__global__ __launch_bounds__(128) void k_final(const float* __restrict__ xn,
                                               const float* __restrict__ yn,
                                               const float* __restrict__ muA,
                                               const float* __restrict__ stdA,
                                               const int* __restrict__ leadA,
                                               const int* __restrict__ shSelA,
                                               const float* __restrict__ sgnA,
                                               const float* __restrict__ filt,
                                               const float* __restrict__ c96,
                                               const float* __restrict__ s96,
                                               const float* __restrict__ mwr,
                                               const float* __restrict__ mwi,
                                               const float* __restrict__ mbr,
                                               const float* __restrict__ mbi,
                                               float* __restrict__ out) {
    int bc = blockIdx.x;
    int b = bc >> 7, c = bc & 127;
    int tid = threadIdx.x;
    __shared__ float yn_s[96], seq_s[96];
    __shared__ float yfR[F2], yfI[F2], ssR[F2], ssI[F2], sdR[F2], sdI[F2];
    __shared__ float yffR[F2], yffI[F2], oR[F2], oI[F2];

    for (int h = tid; h < HDIM; h += 128) yn_s[h] = yn[(size_t)bc * HDIM + h];
    __syncthreads();
    if (tid < F2) {
        float ar = 0.0f, ai = 0.0f;
        for (int h = 0; h < HDIM; ++h) {
            float v = yn_s[h];
            ar += v * c96[tid * HDIM + h];
            ai -= v * s96[tid * HDIM + h];
        }
        yfR[tid] = ar; yfI[tid] = ai;
        ssR[tid] = 0.0f; ssI[tid] = 0.0f; sdR[tid] = 0.0f; sdI[tid] = 0.0f;
    }
    __syncthreads();

    for (int k = 0; k < 8; ++k) {
        int   lid = leadA[bc * 8 + k];
        int   sh  = shSelA[bc * 8 + k];
        float sg  = sgnA[bc * 8 + k];
        for (int h = tid; h < HDIM; h += 128) {
            int idx = LDIM - sh + h;                  // window into concat(xn, yn)
            float v = (idx < LDIM) ? xn[(size_t)(b * 128 + lid) * LDIM + idx]
                                   : yn[(size_t)(b * 128 + lid) * HDIM + (idx - LDIM)];
            seq_s[h] = v * sg;
        }
        __syncthreads();
        if (tid < F2) {
            float ar = 0.0f, ai = 0.0f;
            for (int h = 0; h < HDIM; ++h) {
                float v = seq_s[h];
                ar += v * c96[tid * HDIM + h];
                ai -= v * s96[tid * HDIM + h];
            }
            float fk  = filt[(size_t)bc * OUTF + k * F2 + tid];
            float fdk = filt[(size_t)bc * OUTF + (8 + k) * F2 + tid];
            ar *= fk; ai *= fk;
            ssR[tid] += ar; ssI[tid] += ai;
            sdR[tid] += (ar - yfR[tid]) * fdk;
            sdI[tid] += (ai - yfI[tid]) * fdk;
        }
        __syncthreads();
    }
    if (tid < F2) {
        float fy = filt[(size_t)bc * OUTF + 16 * F2 + tid];
        yffR[tid] = yfR[tid] * fy;
        yffI[tid] = yfI[tid] * fy;
    }
    __syncthreads();
    if (tid < F2) {
        int o = tid;
        float orr = mbr[o], oii = mbi[o];
        for (int f = 0; f < F2; ++f) {
            float wr = mwr[o * 147 + f], wi = mwi[o * 147 + f];
            orr += ssR[f] * wr - ssI[f] * wi;
            oii += ssR[f] * wi + ssI[f] * wr;
        }
        for (int f = 0; f < F2; ++f) {
            float wr = mwr[o * 147 + 49 + f], wi = mwi[o * 147 + 49 + f];
            orr += sdR[f] * wr - sdI[f] * wi;
            oii += sdR[f] * wi + sdI[f] * wr;
        }
        for (int f = 0; f < F2; ++f) {
            float wr = mwr[o * 147 + 98 + f], wi = mwi[o * 147 + 98 + f];
            orr += yffR[f] * wr - yffI[f] * wi;
            oii += yffR[f] * wi + yffI[f] * wr;
        }
        oR[o] = orr; oI[o] = oii;
    }
    __syncthreads();
    float mu = muA[bc], sd = stdA[bc];
    for (int h = tid; h < HDIM; h += 128) {
        float acc = oR[0];
        for (int f = 1; f < 48; ++f)
            acc += 2.0f * (oR[f] * c96[f * HDIM + h] - oI[f] * s96[f * HDIM + h]);
        acc += oR[48] * c96[48 * HDIM + h] - oI[48] * s96[48 * HDIM + h];
        acc *= (1.0f / HDIM);
        float yv = (yn_s[h] + acc) * sd + mu;
        out[((size_t)b * HDIM + h) * 128 + c] = yv;      // output (B,H,C)
    }
}

// ---------------------------------------------------------------------------
// Host launcher
// ---------------------------------------------------------------------------
extern "C" void kernel_launch(void* const* d_in, const int* in_sizes, int n_in,
                              void* d_out, int out_size, void* d_ws, size_t ws_size,
                              hipStream_t stream) {
    (void)in_sizes; (void)n_in; (void)out_size; (void)ws_size;
    const float* x    = (const float*)d_in[0];
    const float* yh   = (const float*)d_in[1];
    const float* temp = (const float*)d_in[2];
    const float* clsw = (const float*)d_in[3];
    const float* bst  = (const float*)d_in[4];
    const float* sbia = (const float*)d_in[5];
    const float* mhw  = (const float*)d_in[6];
    const float* mhb  = (const float*)d_in[7];
    const float* mwr  = (const float*)d_in[8];
    const float* mwi  = (const float*)d_in[9];
    const float* mbr  = (const float*)d_in[10];
    const float* mbi  = (const float*)d_in[11];
    float* out = (float*)d_out;
    char*  ws  = (char*)d_ws;

    size_t off = 0;
    auto alloc = [&](size_t bytes) { size_t r = off; off = (off + bytes + 255) & ~(size_t)255; return r; };
    size_t o_xn   = alloc((size_t)NB * LDIM * 4);
    size_t o_yn   = alloc((size_t)NB * HDIM * 4);
    size_t o_mu   = alloc((size_t)NB * 4);
    size_t o_std  = alloc((size_t)NB * 4);
    size_t o_xnb  = alloc((size_t)NB * KPAD * 2);
    size_t o_BfT  = alloc((size_t)KPAD * KPAD * 2);
    size_t o_RF   = alloc((size_t)NB * KPAD * 4);
    size_t o_BiT  = alloc((size_t)LDIM * KPAD * 2);
    size_t o_cmax = alloc((size_t)16 * 128 * 128 * 4);
    size_t o_rval = alloc((size_t)16 * 128 * 128 * 4);
    size_t o_shf  = alloc((size_t)16 * 128 * 128 * 4);
    size_t o_lead = alloc((size_t)NB * 8 * 4);
    size_t o_shS  = alloc((size_t)NB * 8 * 4);
    size_t o_sgn  = alloc((size_t)NB * 8 * 4);
    size_t o_cf   = alloc((size_t)NB * 8 * 4);
    size_t o_p    = alloc((size_t)NB * 8 * 4);
    size_t o_filt = alloc((size_t)NB * OUTF * 4);
    size_t o_c96  = alloc((size_t)F2 * HDIM * 4);
    size_t o_s96  = alloc((size_t)F2 * HDIM * 4);

    __bf16* BfT = (__bf16*)(ws + o_BfT);
    __bf16* BiT = (__bf16*)(ws + o_BiT);
    __bf16* xnb = (__bf16*)(ws + o_xnb);
    float*  xnf = (float*)(ws + o_xn);
    float*  ynf = (float*)(ws + o_yn);
    float*  muA = (float*)(ws + o_mu);
    float*  sdA = (float*)(ws + o_std);
    float*  RF  = (float*)(ws + o_RF);
    float*  cmx = (float*)(ws + o_cmax);
    float*  rvl = (float*)(ws + o_rval);
    int*    shf = (int*)(ws + o_shf);
    int*    led = (int*)(ws + o_lead);
    int*    shS = (int*)(ws + o_shS);
    float*  sgn = (float*)(ws + o_sgn);
    float*  cfA = (float*)(ws + o_cf);
    float*  pA  = (float*)(ws + o_p);
    float*  flt = (float*)(ws + o_filt);
    float*  c96 = (float*)(ws + o_c96);
    float*  s96 = (float*)(ws + o_s96);

    k_tables<<<4096, 256, 0, stream>>>(BfT, BiT, c96, s96);
    k_norm  <<<NB, 256, 0, stream>>>(x, yh, xnf, ynf, xnb, muA, sdA);
    k_rfft  <<<NB / 16, 256, 0, stream>>>(xnb, BfT, RF);
    k_corr  <<<dim3(8, 128, 16), 128, 0, stream>>>(RF, BiT, cmx, rvl, shf);
    k_topk  <<<NB, 32, 0, stream>>>(cmx, rvl, shf, temp, led, shS, sgn, cfA);
    k_p     <<<NB, 128, 0, stream>>>(x, clsw, bst, sbia, pA);
    k_filt  <<<NB, 256, 0, stream>>>(pA, cfA, mhw, mhb, flt);
    k_final <<<NB, 128, 0, stream>>>(xnf, ynf, muA, sdA, led, shS, sgn, flt,
                                     c96, s96, mwr, mwi, mbr, mbi, out);
}
